// PrunableGPT2Attention_23287312678946
// MI455X (gfx1250) — matrix operand order repaired and checked by reading, not compiled, not run on hardware
//
#include <hip/hip_runtime.h>
#include <hip/hip_bf16.h>

// ---------------------------------------------------------------------------
// GPT-2 attention block for MI455X (gfx1250, wave32, WMMA bf16 16x16x32).
// Both GEMMs + flash attention run on v_wmma_f32_16x16x32_bf16 with
// fp32 accumulation; fp32 inputs are demoted to bf16 once up front.
// GEMM tiles stream through LDS with the CDNA5 async global->LDS path
// (ASYNCcnt) and double buffering.
// ---------------------------------------------------------------------------

typedef __attribute__((ext_vector_type(16))) __bf16 v16bf;
typedef __attribute__((ext_vector_type(8)))  __bf16 v8bf;
typedef __attribute__((ext_vector_type(8)))  float  v8f;

union F8   { v8f v; float f[8]; };
union Frag { v16bf v; v8bf h[2]; };

// Problem sizes
constexpr int NB   = 2;      // batch
constexpr int SEQ  = 2048;   // sequence
constexpr int EMB  = 1024;   // embed
constexpr int NH   = 16;     // heads
constexpr int HD   = 64;     // head dim
constexpr int MROWS = NB * SEQ;   // 4096
constexpr int KDIM  = EMB;        // 1024 (contraction for both GEMMs)

// GEMM tiling
constexpr int BM = 128, BN = 128, BK = 32;
constexpr int LDT = BK + 8;  // padded LDS row stride (halfwords) -> conflict-free

// ---------------------------------------------------------------------------
// CDNA5 async global->LDS copy (ASYNCcnt), per-lane 16B.
// LDS operand is the 32-bit LDS byte address = low word of the generic
// pointer (flat LDS aperture truncates to addr[31:0], ISA 10.2).
// ---------------------------------------------------------------------------
__device__ __forceinline__ void async_copy_b128(void* lds_ptr, const void* gptr) {
  const unsigned lds_addr = (unsigned)(uintptr_t)lds_ptr;
  const unsigned long long gaddr = (unsigned long long)(uintptr_t)gptr;
  asm volatile("global_load_async_to_lds_b128 %0, %1, off"
               :: "v"(lds_addr), "v"(gaddr) : "memory");
}
__device__ __forceinline__ void wait_async0() {
  asm volatile("s_wait_asynccnt 0x0" ::: "memory");
}

// ---------------------------------------------------------------------------
// WMMA fragment loaders (layouts per CDNA5 ISA §7.12.2)
// ---------------------------------------------------------------------------

// A-matrix 16x32 bf16: lane = row (0..15), lanes>=16 same rows;
// elements 0..7 = K {klo..klo+7}, elements 8..15 = K {16+klo..}, klo = 8*(lane>=16)
__device__ __forceinline__ v16bf load_frag_a(const __bf16* base, int ld) {
  const int lane = threadIdx.x & 31;
  const int row  = lane & 15;
  const int klo  = (lane >> 4) << 3;     // 0 or 8
  Frag u;
  u.h[0] = *(const v8bf*)(base + (size_t)row * ld + klo);
  u.h[1] = *(const v8bf*)(base + (size_t)row * ld + klo + 16);
  return u.v;
}

// B-matrix 32x16 bf16, stored K-contiguous per column (i.e. base is B^T row-major):
// lane = column (0..15); elements 0..7 = K {klo..klo+7}, 8..15 = K {16+klo..}, klo = 8*(lane>=16).
__device__ __forceinline__ v16bf load_frag_b(const __bf16* base, int ld) {
  const int lane = threadIdx.x & 31;
  const int col  = lane & 15;
  const int klo  = (lane >> 4) << 3;     // 0 or 8
  Frag u;
  u.h[0] = *(const v8bf*)(base + (size_t)col * ld + klo);
  u.h[1] = *(const v8bf*)(base + (size_t)col * ld + klo + 16);
  return u.v;
}

__device__ __forceinline__ v8f wmma_bf16(v16bf a, v16bf b, v8f c) {
  return __builtin_amdgcn_wmma_f32_16x16x32_bf16(false, a, false, b,
                                                 (short)0, c, false, false);
}

// ---------------------------------------------------------------------------
// Precision demotion kernels
// ---------------------------------------------------------------------------
__global__ void convert_f32_bf16(const float* __restrict__ in,
                                 __bf16* __restrict__ out, int n) {
  int i = blockIdx.x * blockDim.x + threadIdx.x;
  const int stride = gridDim.x * blockDim.x;
  for (; i < n; i += stride) out[i] = (__bf16)in[i];
}

// in [rows][cols] fp32  ->  out [cols][rows] bf16
__global__ void transpose_f32_bf16(const float* __restrict__ in,
                                   __bf16* __restrict__ out,
                                   int rows, int cols) {
  const int c = blockIdx.x * 16 + threadIdx.x;
  const int r = blockIdx.y * 16 + threadIdx.y;
  if (r < rows && c < cols)
    out[(size_t)c * rows + r] = (__bf16)in[(size_t)r * cols + c];
}

// ---------------------------------------------------------------------------
// Shared GEMM core: 256 threads = 8 waves; 128x128 block tile, BK=32;
// wave tile 64x32 -> 8 wmma / K-step; double-buffered async LDS fills.
// ---------------------------------------------------------------------------
#define GEMM_MAIN(ACC)                                                         \
  const int m0 = blockIdx.y * BM;                                              \
  const int n0 = blockIdx.x * BN;                                              \
  const int w    = threadIdx.x >> 5;                                           \
  const int lane = threadIdx.x & 31;                                           \
  const int wm = (w >> 2) * 64;                                                \
  const int wn = (w & 3) * 32;                                                 \
  F8 ACC[4][2] = {};                                                           \
  const int fr = threadIdx.x >> 2;            /* fill row 0..63 (x2) */        \
  const int fc = (threadIdx.x & 3) << 3;      /* fill col 0,8,16,24  */        \
  /* prefill buffer 0 */                                                       \
  {                                                                            \
    async_copy_b128(&As[0][fr * LDT + fc],                                     \
                    &A [(size_t)(m0 + fr) * KDIM + fc]);                       \
    async_copy_b128(&As[0][(fr + 64) * LDT + fc],                              \
                    &A [(size_t)(m0 + fr + 64) * KDIM + fc]);                  \
    async_copy_b128(&Bs[0][fr * LDT + fc],                                     \
                    &Bt[(size_t)(n0 + fr) * KDIM + fc]);                       \
    async_copy_b128(&Bs[0][(fr + 64) * LDT + fc],                              \
                    &Bt[(size_t)(n0 + fr + 64) * KDIM + fc]);                  \
  }                                                                            \
  wait_async0();                                                               \
  __syncthreads();                                                             \
  for (int k0 = 0; k0 < KDIM; k0 += BK) {                                      \
    const int cur = (k0 >> 5) & 1;                                             \
    const int nxt = cur ^ 1;                                                   \
    if (k0 + BK < KDIM) {  /* stream next tile while computing this one */     \
      const int kn = k0 + BK;                                                  \
      async_copy_b128(&As[nxt][fr * LDT + fc],                                 \
                      &A [(size_t)(m0 + fr) * KDIM + kn + fc]);                \
      async_copy_b128(&As[nxt][(fr + 64) * LDT + fc],                          \
                      &A [(size_t)(m0 + fr + 64) * KDIM + kn + fc]);           \
      async_copy_b128(&Bs[nxt][fr * LDT + fc],                                 \
                      &Bt[(size_t)(n0 + fr) * KDIM + kn + fc]);                \
      async_copy_b128(&Bs[nxt][(fr + 64) * LDT + fc],                          \
                      &Bt[(size_t)(n0 + fr + 64) * KDIM + kn + fc]);           \
    }                                                                          \
    v16bf af[4], bf[2];                                                        \
    _Pragma("unroll")                                                          \
    for (int i = 0; i < 4; ++i)                                                \
      af[i] = load_frag_a(&As[cur][(wm + i * 16) * LDT], LDT);                 \
    _Pragma("unroll")                                                          \
    for (int j = 0; j < 2; ++j)                                                \
      bf[j] = load_frag_b(&Bs[cur][(wn + j * 16) * LDT], LDT);                 \
    _Pragma("unroll")                                                          \
    for (int i = 0; i < 4; ++i)                                                \
      _Pragma("unroll")                                                        \
      for (int j = 0; j < 2; ++j)                                              \
        ACC[i][j].v = wmma_bf16(af[i], bf[j], ACC[i][j].v);                    \
    wait_async0();    /* next tile resident */                                 \
    __syncthreads();  /* all waves done reading cur & see nxt */               \
  }

// ---------------------------------------------------------------------------
// QKV GEMM: [4096 x 1024] x [1024 x 3072] (+bias), scatter into
// Q [B,H,S,D], K [B,H,S,D], Vt [B,H,D,S] as bf16.
// ---------------------------------------------------------------------------
__global__ void __launch_bounds__(256)
gemm_qkv_kernel(const __bf16* __restrict__ A,   // Xbf  [4096][1024]
                const __bf16* __restrict__ Bt,  // Wqkv^T [3072][1024]
                const float*  __restrict__ bias,
                __bf16* __restrict__ Qo, __bf16* __restrict__ Ko,
                __bf16* __restrict__ Vto) {
  __shared__ __attribute__((aligned(16))) __bf16 As[2][BM * LDT];
  __shared__ __attribute__((aligned(16))) __bf16 Bs[2][BN * LDT];

  GEMM_MAIN(acc)

  // epilogue: bias + bf16 + head-split scatter (V transposed)
  const int rhalf = (lane >> 4) << 3;
#pragma unroll
  for (int i = 0; i < 4; ++i) {
#pragma unroll
    for (int j = 0; j < 2; ++j) {
      const int col = n0 + wn + j * 16 + (lane & 15);   // 0..3071
      const float bs = bias[col];
#pragma unroll
      for (int r = 0; r < 8; ++r) {
        const int row = m0 + wm + i * 16 + r + rhalf;   // 0..4095
        const float val = acc[i][j].f[r] + bs;
        const __bf16 bv = (__bf16)val;
        const int b = row >> 11;           // /SEQ
        const int s = row & (SEQ - 1);
        if (col < EMB) {
          const int h = col >> 6, d = col & 63;
          Qo[(((size_t)(b * NH + h)) * SEQ + s) * HD + d] = bv;
        } else if (col < 2 * EMB) {
          const int e = col - EMB, h = e >> 6, d = e & 63;
          Ko[(((size_t)(b * NH + h)) * SEQ + s) * HD + d] = bv;
        } else {
          const int e = col - 2 * EMB, h = e >> 6, d = e & 63;
          Vto[(((size_t)(b * NH + h)) * HD + d) * SEQ + s] = bv;   // transposed
        }
      }
    }
  }
}

// ---------------------------------------------------------------------------
// Output projection: [4096 x 1024] x [1024 x 1024] (+bias) -> fp32 d_out
// ---------------------------------------------------------------------------
__global__ void __launch_bounds__(256)
gemm_proj_kernel(const __bf16* __restrict__ A,   // Obf [4096][1024]
                 const __bf16* __restrict__ Bt,  // Wproj^T [1024][1024]
                 const float*  __restrict__ bias,
                 float* __restrict__ Out) {
  __shared__ __attribute__((aligned(16))) __bf16 As[2][BM * LDT];
  __shared__ __attribute__((aligned(16))) __bf16 Bs[2][BN * LDT];

  GEMM_MAIN(acc)

  const int rhalf = (lane >> 4) << 3;
#pragma unroll
  for (int i = 0; i < 4; ++i)
#pragma unroll
    for (int j = 0; j < 2; ++j) {
      const int col = n0 + wn + j * 16 + (lane & 15);
      const float bs = bias[col];
#pragma unroll
      for (int r = 0; r < 8; ++r) {
        const int row = m0 + wm + i * 16 + r + rhalf;
        Out[(size_t)row * EMB + col] = acc[i][j].f[r] + bs;
      }
    }
}

// ---------------------------------------------------------------------------
// Causal flash attention. Grid (S/128, B*H); 8 waves/WG; each wave owns a
// 16-row Q tile, streams 32-key blocks, online softmax in fp32.
// ---------------------------------------------------------------------------
__global__ void __launch_bounds__(256)
attn_kernel(const __bf16* __restrict__ Qg,   // [B,H,S,D]
            const __bf16* __restrict__ Kg,   // [B,H,S,D]
            const __bf16* __restrict__ Vtg,  // [B,H,D,S]
            __bf16* __restrict__ Og) {       // [B,S,E]
  __shared__ __attribute__((aligned(16))) __bf16 Plds[8 * 16 * 40];

  const int lane = threadIdx.x & 31;
  const int w    = threadIdx.x >> 5;
  const int bh = blockIdx.y;
  const int b = bh >> 4, h = bh & 15;
  const __bf16* Qp = Qg  + (size_t)bh * SEQ * HD;
  const __bf16* Kp = Kg  + (size_t)bh * SEQ * HD;
  const __bf16* Vp = Vtg + (size_t)bh * HD * SEQ;
  const int q0 = blockIdx.x * 128 + w * 16;
  const int rhalf = (lane >> 4) << 3;

  // Q fragments (d 0..31, 32..63) direct from global, reused for all K blocks
  v16bf qf[2];
#pragma unroll
  for (int ks = 0; ks < 2; ++ks)
    qf[ks] = load_frag_a(Qp + (size_t)q0 * HD + ks * 32, HD);

  F8 o[4] = {};
  float mrow[8], lrow[8], corr[8];
#pragma unroll
  for (int r = 0; r < 8; ++r) { mrow[r] = -3.0e38f; lrow[r] = 0.0f; }

  __bf16* pw = &Plds[w * 16 * 40];

  for (int kb = 0; kb < q0 + 16; kb += 32) {     // causal pruning
    // S = Q K^T  (two 16-col tiles, two 32-deep k-steps)
    F8 sc[2] = {};
#pragma unroll
    for (int j = 0; j < 2; ++j)
#pragma unroll
      for (int ks = 0; ks < 2; ++ks) {
        v16bf kf = load_frag_b(Kp + (size_t)(kb + j * 16) * HD + ks * 32, HD);
        sc[j].v = wmma_bf16(qf[ks], kf, sc[j].v);
      }

    // scale 1/sqrt(64) + causal mask
#pragma unroll
    for (int j = 0; j < 2; ++j) {
      const int kc = kb + j * 16 + (lane & 15);
#pragma unroll
      for (int r = 0; r < 8; ++r) {
        const int qr = q0 + r + rhalf;
        const float v = sc[j].f[r] * 0.125f;
        sc[j].f[r] = (kc <= qr) ? v : -3.0e38f;
      }
    }

    // online softmax: row reductions stay within 16-lane halves (xor 1/2/4/8)
#pragma unroll
    for (int r = 0; r < 8; ++r) {
      float mx = fmaxf(sc[0].f[r], sc[1].f[r]);
#pragma unroll
      for (int off = 8; off >= 1; off >>= 1) mx = fmaxf(mx, __shfl_xor(mx, off, 32));
      const float mnew = fmaxf(mrow[r], mx);
      const float c = __expf(mrow[r] - mnew);
      mrow[r] = mnew;
      const float p0 = __expf(sc[0].f[r] - mnew);
      const float p1 = __expf(sc[1].f[r] - mnew);
      sc[0].f[r] = p0; sc[1].f[r] = p1;
      float rs = p0 + p1;
#pragma unroll
      for (int off = 8; off >= 1; off >>= 1) rs += __shfl_xor(rs, off, 32);
      lrow[r] = lrow[r] * c + rs;
      corr[r] = c;
    }
#pragma unroll
    for (int t = 0; t < 4; ++t)
#pragma unroll
      for (int r = 0; r < 8; ++r) o[t].f[r] *= corr[r];

    // P: C/D layout -> A layout via per-wave LDS slab
#pragma unroll
    for (int j = 0; j < 2; ++j)
#pragma unroll
      for (int r = 0; r < 8; ++r)
        pw[(r + rhalf) * 40 + j * 16 + (lane & 15)] = (__bf16)sc[j].f[r];
    asm volatile("s_wait_dscnt 0x0" ::: "memory");   // cross-lane LDS RAW (CDNA5 split cnt)

    const v16bf pf = load_frag_a(pw, 40);
#pragma unroll
    for (int t = 0; t < 4; ++t) {
      v16bf vf = load_frag_b(Vp + (size_t)(t * 16) * SEQ + kb, SEQ);
      o[t].v = wmma_bf16(pf, vf, o[t].v);
    }
  }

  // normalize + merge heads -> O [B,S,E] bf16
#pragma unroll
  for (int t = 0; t < 4; ++t)
#pragma unroll
    for (int r = 0; r < 8; ++r) {
      const int qr = q0 + r + rhalf;
      const float val = o[t].f[r] / lrow[r];
      Og[((size_t)b * SEQ + qr) * EMB + h * HD + t * 16 + (lane & 15)] = (__bf16)val;
    }
}

// ---------------------------------------------------------------------------
// Host launcher
// ---------------------------------------------------------------------------
extern "C" void kernel_launch(void* const* d_in, const int* in_sizes, int n_in,
                              void* d_out, int out_size, void* d_ws, size_t ws_size,
                              hipStream_t stream) {
  const float* hs    = (const float*)d_in[0];   // [B,S,E]
  const float* wqkv  = (const float*)d_in[1];   // [E,3E]
  const float* bqkv  = (const float*)d_in[2];   // [3E]
  const float* wproj = (const float*)d_in[3];   // [E,E]
  const float* bproj = (const float*)d_in[4];   // [E]
  float* out = (float*)d_out;

  char* ws = (char*)d_ws;
  size_t off = 0;
  __bf16* Xbf   = (__bf16*)(ws + off); off += (size_t)MROWS * EMB * 2;       // 8 MB
  __bf16* Wqkvt = (__bf16*)(ws + off); off += (size_t)3 * EMB * EMB * 2;     // 6 MB
  __bf16* Wprjt = (__bf16*)(ws + off); off += (size_t)EMB * EMB * 2;         // 2 MB
  __bf16* Qb    = (__bf16*)(ws + off); off += (size_t)MROWS * EMB * 2;       // 8 MB
  __bf16* Kb    = (__bf16*)(ws + off); off += (size_t)MROWS * EMB * 2;       // 8 MB
  __bf16* Vtb   = (__bf16*)(ws + off); off += (size_t)MROWS * EMB * 2;       // 8 MB
  __bf16* Ob    = (__bf16*)(ws + off); off += (size_t)MROWS * EMB * 2;       // 8 MB

  // 1) demote inputs to bf16 (weights transposed so B operands are K-contiguous)
  convert_f32_bf16<<<2048, 256, 0, stream>>>(hs, Xbf, MROWS * EMB);
  transpose_f32_bf16<<<dim3(3 * EMB / 16, EMB / 16), dim3(16, 16), 0, stream>>>(
      wqkv, Wqkvt, EMB, 3 * EMB);
  transpose_f32_bf16<<<dim3(EMB / 16, EMB / 16), dim3(16, 16), 0, stream>>>(
      wproj, Wprjt, EMB, EMB);

  // 2) fused QKV projection + head split (V stored transposed)
  gemm_qkv_kernel<<<dim3(3 * EMB / BN, MROWS / BM), 256, 0, stream>>>(
      Xbf, Wqkvt, bqkv, Qb, Kb, Vtb);

  // 3) causal flash attention
  attn_kernel<<<dim3(SEQ / 128, NB * NH), 256, 0, stream>>>(Qb, Kb, Vtb, Ob);

  // 4) output projection -> fp32
  gemm_proj_kernel<<<dim3(EMB / BN, MROWS / BM), 256, 0, stream>>>(
      Ob, Wprjt, bproj, out);
}
